// MultiHeadAttention_15599321219502
// MI455X (gfx1250) — matrix operand order repaired and checked by reading, not compiled, not run on hardware
//
// MI455X (gfx1250) fused multi-head attention, bf16 WMMA / f32 accumulate.
// 3 kernels: QKV projection -> flash attention (online softmax) -> out proj.
// All GEMMs use v_wmma_f32_16x16x32_bf16 (wave32, 16x16 C tiles).
// Attention kernel streams K/V^T chunks with the Tensor Data Mover
// (tensor_load_to_lds + s_wait_tensorcnt) double-buffered under the WMMAs.
// Workspace layout (bf16): Q[32][8][1024][64], K[same], Vt[32][8][64][1024],
// Oattn[32][1024][512]  => 4 * 32MB = 128MB of d_ws.

#include <hip/hip_runtime.h>
#include <hip/hip_bf16.h>
#include <stdint.h>

#define EMB 64
#define HEADS 8
#define BATCH 32
#define SEQ 1024
#define HID 512
#define QKV_COLS 1536
#define QT 64   // query rows per attention block
#define KC 32   // key chunk

typedef __attribute__((ext_vector_type(16))) __bf16        v16bf;
typedef __attribute__((ext_vector_type(8)))  float         v8f;
typedef __attribute__((ext_vector_type(8)))  unsigned int  v8u;
typedef __attribute__((ext_vector_type(4)))  unsigned int  v4u;
typedef __attribute__((ext_vector_type(8)))  int           v8i;
typedef __attribute__((ext_vector_type(4)))  int           v4i;

#if __has_builtin(__builtin_amdgcn_tensor_load_to_lds) && \
    __has_builtin(__builtin_amdgcn_s_wait_tensorcnt)
#define HAVE_TDM 1
#endif

__device__ __forceinline__ v16bf as_bf16x16(v8u u) {
  return __builtin_bit_cast(v16bf, u);
}

// f32 -> bf16, round-to-nearest-even
__device__ __forceinline__ unsigned short f2bf(float f) {
  unsigned int u = __float_as_uint(f);
  u += 0x7FFFu + ((u >> 16) & 1u);
  return (unsigned short)(u >> 16);
}

// A-fragment K offset for 16x16x32 bf16 (ISA 7.12.2: 16-bit A 16x32):
// lanes 0-15 hold K pairs {0..7,16..23}; lanes 16-31 hold {8..15,24..31}
__device__ __forceinline__ int a_k0(int j, int hi) {
  return (j < 4) ? (hi * 8 + 2 * j) : (16 + hi * 8 + 2 * (j - 4));
}
// B-fragment: same K order as A: lanes 0-15 hold {0..7,16..23}; lanes 16-31 hold {8..15,24..31}
__device__ __forceinline__ int b_k0(int j, int hi) { return a_k0(j, hi); }

// Generic pointer to a __shared__ object: LDS aperture | offset, so the low
// 32 bits are the LDS byte address the TDM D# wants (ISA 10.2 aperture map).
__device__ __forceinline__ unsigned lds_off(const void* p) {
  return (unsigned)(unsigned long long)p;
}

// Issue one TDM 2D tile load (bf16 elements). Descriptor per ISA 08 §8:
//  group0: count=1 | lds_addr | global_addr[56:0] | type=2
//  group1: data_size=1(2B), tensor_dim0/1, tile_dim0/1, tensor_dim0_stride
// tile/tensor dims and stride are in elements (data_size units).
__device__ __forceinline__ void tdm_load_2d(const void* gaddr, unsigned lds,
                                            unsigned tile0, unsigned tile1,
                                            unsigned ten0, unsigned ten1,
                                            unsigned stride0) {
#ifdef HAVE_TDM
  unsigned long long ga = (unsigned long long)gaddr;
  v4u g0;
  g0[0] = 1u;                                  // count=1, user descriptor
  g0[1] = lds;                                 // lds_addr (bytes)
  g0[2] = (unsigned)ga;                        // global_addr[31:0]
  g0[3] = (unsigned)((ga >> 32) & 0x01FFFFFFu) // global_addr[56:32]
          | 0x80000000u;                       // type=2 at bits [127:126]
  v8i g1;
  g1[0] = (int)(1u << 16);                     // data_size=1 (2 bytes)
  g1[1] = (int)((ten0 & 0xFFFFu) << 16);       // tensor_dim0[15:0] at [63:48]
  g1[2] = (int)((ten0 >> 16) | ((ten1 & 0xFFFFu) << 16));  // dim0 hi | dim1 lo
  g1[3] = (int)((ten1 >> 16) | (tile0 << 16)); // dim1 hi | tile_dim0
  g1[4] = (int)(tile1 & 0xFFFFu);              // tile_dim1 (tile_dim2=0)
  g1[5] = (int)stride0;                        // tensor_dim0_stride[31:0]
  g1[6] = 0;                                   // stride hi / dim1_stride lo
  g1[7] = 0;
  v4i z = {0, 0, 0, 0};
#if defined(__clang_major__) && __clang_major__ >= 23
  v8i z8 = {0, 0, 0, 0, 0, 0, 0, 0};
  __builtin_amdgcn_tensor_load_to_lds(g0, g1, z, z, z8, 0);
#else
  __builtin_amdgcn_tensor_load_to_lds(g0, g1, z, z, 0);
#endif
#endif
}

// ---------------------------------------------------------------- kernel 1
// QKV projection: rows of x (16 per block) x 128 output cols per block.
// De-interleaves cols (h*192 + d*3 + sel) into Q / K / V^T bf16 buffers.
__global__ __launch_bounds__(256) void qkv_kernel(
    const float* __restrict__ x, const float* __restrict__ w_qkv,
    const float* __restrict__ b_qkv, unsigned short* __restrict__ Qw,
    unsigned short* __restrict__ Kw, unsigned short* __restrict__ Vtw) {
  __shared__ alignas(16) unsigned short xs[16 * 64];    // x tile, bf16
  __shared__ alignas(16) unsigned short wt[128 * 64];   // W tile, [col][k]
  const int tid = threadIdx.x;
  const int lane = tid & 31, wave = tid >> 5;
  const int hi = lane >> 4, ln = lane & 15;
  const int rowbase = blockIdx.x * 16;   // flattened (b*1024+n) row
  const int cb = blockIdx.y * 128;       // output column group

  for (int i = tid; i < 16 * 64; i += 256)
    xs[i] = f2bf(x[(rowbase + (i >> 6)) * 64 + (i & 63)]);
  for (int i = tid; i < 128 * 64; i += 256) {
    int k = i >> 7, c = i & 127;  // coalesced over c
    wt[c * 64 + k] = f2bf(w_qkv[k * QKV_COLS + cb + c]);
  }
  __syncthreads();

  v8u au[2];
#pragma unroll
  for (int s = 0; s < 2; ++s)
#pragma unroll
    for (int j = 0; j < 8; ++j)
      au[s][j] = *(const unsigned int*)&xs[ln * 64 + s * 32 + a_k0(j, hi)];

  v8f c = {};
  const int clocal = wave * 16 + ln;
#pragma unroll
  for (int s = 0; s < 2; ++s) {
    v8u bu;
#pragma unroll
    for (int j = 0; j < 8; ++j)
      bu[j] = *(const unsigned int*)&wt[clocal * 64 + s * 32 + b_k0(j, hi)];
    c = __builtin_amdgcn_wmma_f32_16x16x32_bf16(
        false, as_bf16x16(au[s]), false, as_bf16x16(bu), (short)0, c, false,
        false);
  }

  const int col = cb + clocal;
  const float bias = b_qkv[col];
  const int sel = col % 3, t2 = col / 3;
  const int h = t2 >> 6, dd = t2 & 63;
  const int bb = rowbase >> 10;
  const int nbase = (rowbase & 1023) + hi * 8;
  const int bh = bb * HEADS + h;
#pragma unroll
  for (int j = 0; j < 8; ++j) {
    unsigned short val = f2bf(c[j] + bias);
    int nq = nbase + j;
    if (sel == 0)
      Qw[(bh * SEQ + nq) * EMB + dd] = val;
    else if (sel == 1)
      Kw[(bh * SEQ + nq) * EMB + dd] = val;
    else
      Vtw[(bh * EMB + dd) * SEQ + nq] = val;
  }
}

// ---------------------------------------------------------------- kernel 2
// Flash attention per (b,h): 64 query rows per block, 4 waves x 16 rows.
// K/V^T chunks streamed by the Tensor Data Mover, double-buffered so the
// DMA of chunk i+1 overlaps the 8 WMMAs of chunk i. Online softmax;
// reference divides by sqrt(64)=8 AFTER softmax -> folded into final 1/l.
__global__ __launch_bounds__(128) void attn_kernel(
    const unsigned short* __restrict__ Qw, const unsigned short* __restrict__ Kw,
    const unsigned short* __restrict__ Vtw, unsigned short* __restrict__ Ow) {
  __shared__ alignas(16) unsigned short qs[QT * EMB];        // 8KB
  __shared__ alignas(16) unsigned short kcs[2][KC * EMB];    // 2x4KB [key][d]
  __shared__ alignas(16) unsigned short vcs[2][EMB * KC];    // 2x4KB [d][key]
  __shared__ alignas(16) unsigned short ps[4][16 * KC];      // per-wave P tile
  const int tid = threadIdx.x;
  const int lane = tid & 31, wave = tid >> 5;
  const int hi = lane >> 4, ln = lane & 15;
  const int bh = blockIdx.y;
  const int qbase = blockIdx.x * QT;

  const unsigned short* Qg = Qw + (size_t)bh * SEQ * EMB;
  const unsigned short* Kg = Kw + (size_t)bh * SEQ * EMB;
  const unsigned short* Vg = Vtw + (size_t)bh * EMB * SEQ;

#ifdef HAVE_TDM
  // Prologue: wave 0 DMAs the Q tile and chunk 0 of K / V^T into LDS.
  if (wave == 0) {
    tdm_load_2d(Qg + qbase * EMB, lds_off(qs), QT * EMB, 1, QT * EMB, 1,
                QT * EMB);
    tdm_load_2d(Kg, lds_off(kcs[0]), KC * EMB, 1, KC * EMB, 1, KC * EMB);
    tdm_load_2d(Vg, lds_off(vcs[0]), KC, EMB, SEQ, EMB, SEQ);
    __builtin_amdgcn_s_wait_tensorcnt(0);
  }
  __syncthreads();
#else
  const unsigned int* Qu = (const unsigned int*)Qg;
  unsigned int* qsu = (unsigned int*)qs;
  for (int i = tid; i < QT * EMB / 2; i += 128) qsu[i] = Qu[qbase * 32 + i];
  __syncthreads();
#endif

  // Q fragments held in registers for the whole key sweep
  v8u qa[2];
  const int qrow = wave * 16 + ln;
#pragma unroll
  for (int s = 0; s < 2; ++s)
#pragma unroll
    for (int j = 0; j < 8; ++j)
      qa[s][j] = *(const unsigned int*)&qs[qrow * EMB + s * 32 + a_k0(j, hi)];

  v8f o[4] = {};
  float m[8], l[8];
#pragma unroll
  for (int j = 0; j < 8; ++j) { m[j] = -3.0e38f; l[j] = 0.0f; }

  for (int kb = 0, it = 0; kb < SEQ; kb += KC, ++it) {
    const int cur = it & 1;
#ifdef HAVE_TDM
    // Everyone passed last iteration's barrier => buf[1-cur] is reusable.
    if (wave == 0 && kb + KC < SEQ) {
      tdm_load_2d(Kg + (kb + KC) * EMB, lds_off(kcs[1 - cur]), KC * EMB, 1,
                  KC * EMB, 1, KC * EMB);
      tdm_load_2d(Vg + (kb + KC), lds_off(vcs[1 - cur]), KC, EMB, SEQ, EMB,
                  SEQ);
    }
#else
    __syncthreads();
    {
      const unsigned int* Ku = (const unsigned int*)Kg;
      const unsigned int* Vu = (const unsigned int*)Vg;
      unsigned int* ksu = (unsigned int*)kcs[cur];
      unsigned int* vsu = (unsigned int*)vcs[cur];
      for (int i = tid; i < KC * EMB / 2; i += 128) ksu[i] = Ku[kb * 32 + i];
      for (int i = tid; i < EMB * KC / 2; i += 128) {
        int dd = i >> 4, w = i & 15;
        vsu[dd * 16 + w] = Vu[dd * 512 + (kb >> 1) + w];
      }
    }
    __syncthreads();
    if (kb + KC < SEQ) {
      const unsigned int* Ku = (const unsigned int*)Kg;
      __builtin_prefetch(&Ku[(kb + KC) * 32 + tid * 8], 0, 0);
    }
#endif

    // S(16x32) = Q(16x64) * K^T : two 16x16 C tiles, two k-steps
    v8f s0 = {}, s1 = {};
#pragma unroll
    for (int s = 0; s < 2; ++s) {
      v8u kf0, kf1;
#pragma unroll
      for (int j = 0; j < 8; ++j) {
        int k0 = s * 32 + b_k0(j, hi);
        kf0[j] = *(const unsigned int*)&kcs[cur][ln * EMB + k0];
        kf1[j] = *(const unsigned int*)&kcs[cur][(16 + ln) * EMB + k0];
      }
      s0 = __builtin_amdgcn_wmma_f32_16x16x32_bf16(
          false, as_bf16x16(qa[s]), false, as_bf16x16(kf0), (short)0, s0,
          false, false);
      s1 = __builtin_amdgcn_wmma_f32_16x16x32_bf16(
          false, as_bf16x16(qa[s]), false, as_bf16x16(kf1), (short)0, s1,
          false, false);
    }

    // Online softmax: row r = hi*8 + j lives in VGPR j across one 16-lane half
#pragma unroll
    for (int j = 0; j < 8; ++j) {
      float rv = fmaxf(s0[j], s1[j]);
      rv = fmaxf(rv, __shfl_xor(rv, 1));
      rv = fmaxf(rv, __shfl_xor(rv, 2));
      rv = fmaxf(rv, __shfl_xor(rv, 4));
      rv = fmaxf(rv, __shfl_xor(rv, 8));
      float mn = fmaxf(m[j], rv);
      float sc = __expf(m[j] - mn);
      m[j] = mn;
      float p0 = __expf(s0[j] - mn);
      float p1 = __expf(s1[j] - mn);
      s0[j] = p0;
      s1[j] = p1;
      float rs = p0 + p1;
      rs += __shfl_xor(rs, 1);
      rs += __shfl_xor(rs, 2);
      rs += __shfl_xor(rs, 4);
      rs += __shfl_xor(rs, 8);
      l[j] = l[j] * sc + rs;
#pragma unroll
      for (int dt = 0; dt < 4; ++dt) o[dt][j] *= sc;
    }

    // C-layout -> A-layout relayout of P through per-wave LDS tile
#pragma unroll
    for (int j = 0; j < 8; ++j) {
      int r = hi * 8 + j;
      ps[wave][r * KC + ln] = f2bf(s0[j]);
      ps[wave][r * KC + 16 + ln] = f2bf(s1[j]);
    }
    v8u pa;
#pragma unroll
    for (int j = 0; j < 8; ++j)
      pa[j] = *(const unsigned int*)&ps[wave][ln * KC + a_k0(j, hi)];

    // O(16x64) += P(16x32) * V(32x64): 4 d-tiles, single k-step
#pragma unroll
    for (int dt = 0; dt < 4; ++dt) {
      v8u vf;
#pragma unroll
      for (int j = 0; j < 8; ++j)
        vf[j] = *(const unsigned int*)&vcs[cur][(dt * 16 + ln) * KC + b_k0(j, hi)];
      o[dt] = __builtin_amdgcn_wmma_f32_16x16x32_bf16(
          false, as_bf16x16(pa), false, as_bf16x16(vf), (short)0, o[dt], false,
          false);
    }

#ifdef HAVE_TDM
    if (wave == 0) __builtin_amdgcn_s_wait_tensorcnt(0);  // next chunk landed
    __syncthreads();  // all waves done with buf[cur]; next chunk visible
#endif
  }

  // epilogue: softmax normalization and the reference's post-softmax /8
  const int b = bh >> 3, h = bh & 7;
#pragma unroll
  for (int j = 0; j < 8; ++j) {
    float inv = 1.0f / (l[j] * 8.0f);
    int n = qbase + wave * 16 + hi * 8 + j;
    size_t rowoff = ((size_t)b * SEQ + n) * HID + h * EMB;
#pragma unroll
    for (int dt = 0; dt < 4; ++dt)
      Ow[rowoff + dt * 16 + ln] = f2bf(o[dt][j] * inv);
  }
}

// ---------------------------------------------------------------- kernel 3
// Output projection: (32768 x 512) @ (512 x 64) + bias, fp32 out.
__global__ __launch_bounds__(128) void proj_kernel(
    const unsigned short* __restrict__ Ow, const float* __restrict__ w_proj,
    const float* __restrict__ b_proj, float* __restrict__ out) {
  __shared__ alignas(16) unsigned short as_[16 * 64];  // A chunk
  __shared__ alignas(16) unsigned short wt[64 * 64];   // W chunk, [col][k]
  const int tid = threadIdx.x;
  const int lane = tid & 31, wave = tid >> 5;
  const int hi = lane >> 4, ln = lane & 15;
  const int rowbase = blockIdx.x * 16;
  const unsigned int* Au = (const unsigned int*)Ow;
  unsigned int* asu = (unsigned int*)as_;

  v8f c = {};
  for (int kb = 0; kb < HID; kb += 64) {
    __syncthreads();
    for (int i = tid; i < 16 * 32; i += 128) {
      int r = i >> 5, w = i & 31;
      asu[r * 32 + w] = Au[(size_t)(rowbase + r) * 256 + (kb >> 1) + w];
    }
    for (int i = tid; i < 64 * 64; i += 128) {
      int k = i >> 6, cc = i & 63;  // coalesced over cc
      wt[cc * 64 + k] = f2bf(w_proj[(kb + k) * EMB + cc]);
    }
    __syncthreads();
#pragma unroll
    for (int s = 0; s < 2; ++s) {
      v8u au, bu;
#pragma unroll
      for (int j = 0; j < 8; ++j) {
        au[j] = *(const unsigned int*)&as_[ln * 64 + s * 32 + a_k0(j, hi)];
        bu[j] = *(const unsigned int*)&wt[(wave * 16 + ln) * 64 + s * 32 + b_k0(j, hi)];
      }
      c = __builtin_amdgcn_wmma_f32_16x16x32_bf16(
          false, as_bf16x16(au), false, as_bf16x16(bu), (short)0, c, false,
          false);
    }
  }

  const int col = wave * 16 + ln;
  const float bias = b_proj[col];
#pragma unroll
  for (int j = 0; j < 8; ++j) {
    int r = rowbase + hi * 8 + j;
    out[(size_t)r * EMB + col] = c[j] + bias;
  }
}

// ---------------------------------------------------------------- launch
extern "C" void kernel_launch(void* const* d_in, const int* in_sizes, int n_in,
                              void* d_out, int out_size, void* d_ws,
                              size_t ws_size, hipStream_t stream) {
  const float* x = (const float*)d_in[0];
  const float* w_qkv = (const float*)d_in[1];
  const float* b_qkv = (const float*)d_in[2];
  const float* w_proj = (const float*)d_in[3];
  const float* b_proj = (const float*)d_in[4];
  float* out = (float*)d_out;

  const size_t per = (size_t)BATCH * HEADS * SEQ * EMB;  // 16.7M bf16 elems
  unsigned short* Qw = (unsigned short*)d_ws;
  unsigned short* Kw = Qw + per;
  unsigned short* Vtw = Kw + per;
  unsigned short* Ow = Vtw + per;  // [b][n][h*64] bf16

  qkv_kernel<<<dim3(BATCH * SEQ / 16, QKV_COLS / 128), 256, 0, stream>>>(
      x, w_qkv, b_qkv, Qw, Kw, Vtw);
  attn_kernel<<<dim3(SEQ / QT, BATCH * HEADS), 128, 0, stream>>>(Qw, Kw, Vtw, Ow);
  proj_kernel<<<dim3(BATCH * SEQ / 16), 128, 0, stream>>>(Ow, w_proj, b_proj, out);
}
